// SERECalibSparseMoeBlock_89309549953419
// MI455X (gfx1250) — matrix operand order repaired and checked by reading, not compiled, not run on hardware
//
#include <hip/hip_runtime.h>
#include <hip/hip_bf16.h>
#include <stdint.h>

// ---------------------------------------------------------------------------
// Problem constants (from reference): B=2,S=1024 -> T=2048; H=1024; I=2048; E=8
// ---------------------------------------------------------------------------
#define T_TOK 2048
#define HDIM  1024
#define IDIM  2048
#define NEXP  8

typedef __attribute__((ext_vector_type(16))) __bf16 v16bf;
typedef __attribute__((ext_vector_type(8)))  float  v8f;

#define LDT 40  // padded LDS row stride (elements); 80B rows keep 16B alignment

// ---- CDNA5 async LDS DMA (GLOBAL_LOAD_ASYNC_TO_LDS_B128, ASYNCcnt) --------
#if defined(__gfx1250__) && __has_builtin(__builtin_amdgcn_global_load_async_to_lds_b128)
#define HAVE_ASYNC_LDS 1
#else
#define HAVE_ASYNC_LDS 0
#endif

#if HAVE_ASYNC_LDS
typedef int b128_t __attribute__((vector_size(16)));          // GCC-style v4i
typedef __attribute__((address_space(1))) b128_t* glb128_p;   // global src
typedef __attribute__((address_space(3))) b128_t* lds128_p;   // LDS dst
#endif

__device__ __forceinline__ void async_ld16(void* lds, const void* g) {
#if HAVE_ASYNC_LDS
  // Generic LDS address: low 32 bits are the LDS byte offset (aperture rule),
  // so inttoptr into AS(3) yields a valid LDS pointer.
  glb128_p gp = (glb128_p)(uintptr_t)g;                              // drop const
  lds128_p lp = (lds128_p)(uintptr_t)(uint32_t)(uintptr_t)lds;
  __builtin_amdgcn_global_load_async_to_lds_b128(gp, lp, 0, 0);
#else
  (void)lds; (void)g;
#endif
}

__device__ __forceinline__ void wait_async0() {
#if HAVE_ASYNC_LDS
#if __has_builtin(__builtin_amdgcn_s_wait_asynccnt)
  __builtin_amdgcn_s_wait_asynccnt(0);
#else
  asm volatile("s_wait_asynccnt 0x0" ::: "memory");
#endif
#endif
}

// ---- helpers --------------------------------------------------------------
__device__ __forceinline__ unsigned short f2bf(float f) {
  union { float f; uint32_t u; } c; c.f = f;
  uint32_t u = c.u;
  uint32_t r = u + 0x7FFFu + ((u >> 16) & 1u);   // round-to-nearest-even
  return (unsigned short)(r >> 16);
}

__device__ __forceinline__ v16bf load_frag_a(const unsigned short* p) {
  // p = &tile[(mBase + (lane&15))*LDT + (lane>>4)*16]
  // lanes 0-15: K=0..15 ; lanes 16-31: K=16..31 (contiguous, matches B)
  union { v16bf v; uint4 q[2]; } u;
  u.q[0] = *(const uint4*)(p);
  u.q[1] = *(const uint4*)(p + 8);
  return u.v;
}
__device__ __forceinline__ v16bf load_frag_b(const unsigned short* p) {
  // p = &tile[(nBase + (lane&15))*LDT + (lane>>4)*16] ; 16 contiguous K values
  union { v16bf v; uint4 q[2]; } u;
  u.q[0] = *(const uint4*)(p);
  u.q[1] = *(const uint4*)(p + 8);
  return u.v;
}
__device__ __forceinline__ v8f vzero8() {
  v8f z;
  #pragma unroll
  for (int j = 0; j < 8; j++) z[j] = 0.0f;
  return z;
}

// ---------------------------------------------------------------------------
// fp32 -> bf16 conversion (grid-stride)
// ---------------------------------------------------------------------------
__global__ void cvt_f32_bf16(const float* __restrict__ src,
                             unsigned short* __restrict__ dst, int n) {
  int i = blockIdx.x * blockDim.x + threadIdx.x;
  int stride = gridDim.x * blockDim.x;
  for (; i < n; i += stride) dst[i] = f2bf(src[i]);
}

// ---------------------------------------------------------------------------
// Router: logits -> softmax -> top-2 -> normalized weights. One wave per token.
// ---------------------------------------------------------------------------
__global__ void router_kernel(const float* __restrict__ x,
                              const float* __restrict__ gw,
                              float* __restrict__ rw, int* __restrict__ ridx) {
  __shared__ float gws[NEXP * HDIM];  // 32 KB
  for (int i = threadIdx.x; i < NEXP * HDIM; i += blockDim.x) gws[i] = gw[i];
  __syncthreads();
  const int wave = threadIdx.x >> 5, lane = threadIdx.x & 31;
  const int t = blockIdx.x * (blockDim.x >> 5) + wave;
  if (t >= T_TOK) return;
  float acc[NEXP];
  #pragma unroll
  for (int e = 0; e < NEXP; e++) acc[e] = 0.f;
  for (int h = lane; h < HDIM; h += 32) {
    float xv = x[t * HDIM + h];
    #pragma unroll
    for (int e = 0; e < NEXP; e++) acc[e] += xv * gws[e * HDIM + h];
  }
  #pragma unroll
  for (int e = 0; e < NEXP; e++)
    for (int off = 16; off > 0; off >>= 1) acc[e] += __shfl_xor(acc[e], off, 32);
  if (lane == 0) {
    float m = acc[0];
    #pragma unroll
    for (int e = 1; e < NEXP; e++) m = fmaxf(m, acc[e]);
    float p[NEXP], s = 0.f;
    #pragma unroll
    for (int e = 0; e < NEXP; e++) { p[e] = __expf(acc[e] - m); s += p[e]; }
    float inv = 1.f / s;
    #pragma unroll
    for (int e = 0; e < NEXP; e++) p[e] *= inv;
    int i0 = 0;
    for (int e = 1; e < NEXP; e++) if (p[e] > p[i0]) i0 = e;
    int i1 = (i0 == 0) ? 1 : 0;
    for (int e = 0; e < NEXP; e++) if (e != i0 && p[e] > p[i1]) i1 = e;
    float v0 = p[i0], v1 = p[i1];
    float d = fmaxf(v0 + v1, 1e-12f);
    rw[t * 2 + 0] = v0 / d;  rw[t * 2 + 1] = v1 / d;
    ridx[t * 2 + 0] = i0;    ridx[t * 2 + 1] = i1;
  }
}

// ---------------------------------------------------------------------------
// GEMM1 (fused SiLU): inter[e] = silu(x@Wg[e]^T) * (x@Wu[e]^T), bf16 out.
// Block tile: 128(T) x 128(I), BK=32. 8 waves in 2x4; wave tile 64x32 per C.
// Double-buffered LDS fed by GLOBAL_LOAD_ASYNC_TO_LDS_B128 (ASYNCcnt).
// ---------------------------------------------------------------------------
__global__ __launch_bounds__(256, 1)
void gemm1_silu(const unsigned short* __restrict__ xb,
                const unsigned short* __restrict__ w1b,
                unsigned short* __restrict__ interb) {
  __shared__ unsigned short As[2][128 * LDT];
  __shared__ unsigned short Bg[2][128 * LDT];
  __shared__ unsigned short Bu[2][128 * LDT];

  const int e = blockIdx.z;
  const int tRow0 = blockIdx.x * 128;
  const int iCol0 = blockIdx.y * 128;
  const int tid = threadIdx.x;
  const int wave = tid >> 5, lane = tid & 31;
  const int waveM = wave >> 2;   // 0..1 -> 64-row half
  const int waveN = wave & 3;    // 0..3 -> 32-col slice
  const int sel = lane >> 4, l15 = lane & 15;
  const unsigned short* w1g = w1b + (size_t)e * (2 * IDIM) * HDIM;

  v8f accG[4][2], accU[4][2];
  #pragma unroll
  for (int m = 0; m < 4; m++)
    #pragma unroll
    for (int n = 0; n < 2; n++) { accG[m][n] = vzero8(); accU[m][n] = vzero8(); }

#if HAVE_ASYNC_LDS
  auto issue = [&](int kb, int buf) {
    #pragma unroll
    for (int rep = 0; rep < 2; rep++) {
      int c = tid + rep * 256;           // 512 16B chunks per tile
      int row = c >> 2;
      int kc = (c & 3) * 8;
      async_ld16(&As[buf][row * LDT + kc],
                 &xb[(size_t)(tRow0 + row) * HDIM + kb + kc]);
      async_ld16(&Bg[buf][row * LDT + kc],
                 &w1g[(size_t)(iCol0 + row) * HDIM + kb + kc]);
      async_ld16(&Bu[buf][row * LDT + kc],
                 &w1g[(size_t)(IDIM + iCol0 + row) * HDIM + kb + kc]);
    }
  };
  issue(0, 0);
#endif

  for (int kb = 0, it = 0; kb < HDIM; kb += 32, ++it) {
    const int buf = it & 1;
#if HAVE_ASYNC_LDS
    wait_async0();          // this wave's DMA into buf complete
    __syncthreads();        // everyone's DMA into buf visible
    if (kb + 32 < HDIM) issue(kb + 32, buf ^ 1);   // overlap next tile with WMMA
    if (kb + 64 < HDIM) {   // warm caches for tile after next (WGP scope)
      __builtin_prefetch(&xb[(size_t)(tRow0 + (tid >> 1)) * HDIM + kb + 64], 0, 3);
      __builtin_prefetch(&w1g[(size_t)(iCol0 + (tid >> 1)) * HDIM + kb + 64], 0, 3);
    }
#else
    __syncthreads();
    #pragma unroll
    for (int rep = 0; rep < 2; rep++) {
      int c = tid + rep * 256;
      int row = c >> 2;
      int kc = (c & 3) * 8;
      *(uint4*)&As[buf][row * LDT + kc] =
          *(const uint4*)&xb[(size_t)(tRow0 + row) * HDIM + kb + kc];
      *(uint4*)&Bg[buf][row * LDT + kc] =
          *(const uint4*)&w1g[(size_t)(iCol0 + row) * HDIM + kb + kc];
      *(uint4*)&Bu[buf][row * LDT + kc] =
          *(const uint4*)&w1g[(size_t)(IDIM + iCol0 + row) * HDIM + kb + kc];
    }
    __syncthreads();
#endif

    const unsigned short* Ab  = As[buf];
    const unsigned short* Bgb = Bg[buf];
    const unsigned short* Bub = Bu[buf];
    v16bf bg0 = load_frag_b(&Bgb[(waveN * 32 +  0 + l15) * LDT + sel * 16]);
    v16bf bg1 = load_frag_b(&Bgb[(waveN * 32 + 16 + l15) * LDT + sel * 16]);
    v16bf bu0 = load_frag_b(&Bub[(waveN * 32 +  0 + l15) * LDT + sel * 16]);
    v16bf bu1 = load_frag_b(&Bub[(waveN * 32 + 16 + l15) * LDT + sel * 16]);
    #pragma unroll
    for (int m = 0; m < 4; m++) {
      v16bf a = load_frag_a(&Ab[(waveM * 64 + m * 16 + l15) * LDT + sel * 16]);
      accG[m][0] = __builtin_amdgcn_wmma_f32_16x16x32_bf16(false, a, false, bg0,
                      (short)0, accG[m][0], false, false);
      accG[m][1] = __builtin_amdgcn_wmma_f32_16x16x32_bf16(false, a, false, bg1,
                      (short)0, accG[m][1], false, false);
      accU[m][0] = __builtin_amdgcn_wmma_f32_16x16x32_bf16(false, a, false, bu0,
                      (short)0, accU[m][0], false, false);
      accU[m][1] = __builtin_amdgcn_wmma_f32_16x16x32_bf16(false, a, false, bu1,
                      (short)0, accU[m][1], false, false);
    }
  }

  // Epilogue: inter = silu(gate) * up -> bf16
  #pragma unroll
  for (int m = 0; m < 4; m++)
    #pragma unroll
    for (int n = 0; n < 2; n++)
      #pragma unroll
      for (int j = 0; j < 8; j++) {
        int t = tRow0 + waveM * 64 + m * 16 + sel * 8 + j;  // C layout: M=j+sel*8
        int i = iCol0 + waveN * 32 + n * 16 + l15;          //           N=lane&15
        float g = accG[m][n][j];
        float u = accU[m][n][j];
        float s = g * (1.f / (1.f + __expf(-g))) * u;
        interb[((size_t)e * T_TOK + t) * IDIM + i] = f2bf(s);
      }
}

// ---------------------------------------------------------------------------
// GEMM2: expert_out[e] = inter[e] @ down[e]^T  (T x H, fp32 out), K = I = 2048
// ---------------------------------------------------------------------------
__global__ __launch_bounds__(256, 1)
void gemm2(const unsigned short* __restrict__ interb,
           const unsigned short* __restrict__ w2b,
           float* __restrict__ eout) {
  __shared__ unsigned short As[2][128 * LDT];
  __shared__ unsigned short Bs[2][128 * LDT];

  const int e = blockIdx.z;
  const int tRow0 = blockIdx.x * 128;
  const int hCol0 = blockIdx.y * 128;
  const int tid = threadIdx.x;
  const int wave = tid >> 5, lane = tid & 31;
  const int waveM = wave >> 2, waveN = wave & 3;
  const int sel = lane >> 4, l15 = lane & 15;
  const unsigned short* A  = interb + (size_t)e * T_TOK * IDIM;
  const unsigned short* Bw = w2b    + (size_t)e * HDIM * IDIM;

  v8f acc[4][2];
  #pragma unroll
  for (int m = 0; m < 4; m++)
    #pragma unroll
    for (int n = 0; n < 2; n++) acc[m][n] = vzero8();

#if HAVE_ASYNC_LDS
  auto issue = [&](int kb, int buf) {
    #pragma unroll
    for (int rep = 0; rep < 2; rep++) {
      int c = tid + rep * 256;
      int row = c >> 2;
      int kc = (c & 3) * 8;
      async_ld16(&As[buf][row * LDT + kc],
                 &A[(size_t)(tRow0 + row) * IDIM + kb + kc]);
      async_ld16(&Bs[buf][row * LDT + kc],
                 &Bw[(size_t)(hCol0 + row) * IDIM + kb + kc]);
    }
  };
  issue(0, 0);
#endif

  for (int kb = 0, it = 0; kb < IDIM; kb += 32, ++it) {
    const int buf = it & 1;
#if HAVE_ASYNC_LDS
    wait_async0();
    __syncthreads();
    if (kb + 32 < IDIM) issue(kb + 32, buf ^ 1);
    if (kb + 64 < IDIM) {
      __builtin_prefetch(&A [(size_t)(tRow0 + (tid >> 1)) * IDIM + kb + 64], 0, 3);
      __builtin_prefetch(&Bw[(size_t)(hCol0 + (tid >> 1)) * IDIM + kb + 64], 0, 3);
    }
#else
    __syncthreads();
    #pragma unroll
    for (int rep = 0; rep < 2; rep++) {
      int c = tid + rep * 256;
      int row = c >> 2;
      int kc = (c & 3) * 8;
      *(uint4*)&As[buf][row * LDT + kc] =
          *(const uint4*)&A[(size_t)(tRow0 + row) * IDIM + kb + kc];
      *(uint4*)&Bs[buf][row * LDT + kc] =
          *(const uint4*)&Bw[(size_t)(hCol0 + row) * IDIM + kb + kc];
    }
    __syncthreads();
#endif

    const unsigned short* Ab = As[buf];
    const unsigned short* Bb = Bs[buf];
    v16bf b0 = load_frag_b(&Bb[(waveN * 32 +  0 + l15) * LDT + sel * 16]);
    v16bf b1 = load_frag_b(&Bb[(waveN * 32 + 16 + l15) * LDT + sel * 16]);
    #pragma unroll
    for (int m = 0; m < 4; m++) {
      v16bf a = load_frag_a(&Ab[(waveM * 64 + m * 16 + l15) * LDT + sel * 16]);
      acc[m][0] = __builtin_amdgcn_wmma_f32_16x16x32_bf16(false, a, false, b0,
                     (short)0, acc[m][0], false, false);
      acc[m][1] = __builtin_amdgcn_wmma_f32_16x16x32_bf16(false, a, false, b1,
                     (short)0, acc[m][1], false, false);
    }
  }

  #pragma unroll
  for (int m = 0; m < 4; m++)
    #pragma unroll
    for (int n = 0; n < 2; n++)
      #pragma unroll
      for (int j = 0; j < 8; j++) {
        int t = tRow0 + waveM * 64 + m * 16 + sel * 8 + j;
        int h = hCol0 + waveN * 32 + n * 16 + l15;
        eout[((size_t)e * T_TOK + t) * HDIM + h] = acc[m][n][j];
      }
}

// ---------------------------------------------------------------------------
// Gram matrix: upper triangle (36 entries) accumulated in registers,
// wave-reduced, atomicAdd to ws. Zeroed each launch for determinism.
// ---------------------------------------------------------------------------
__global__ void zero36(float* g) { if (threadIdx.x < 64) g[threadIdx.x] = 0.f; }

__global__ void gram_partial(const float* __restrict__ eout,
                             float* __restrict__ gram) {
  const size_t N = (size_t)T_TOK * HDIM;
  float acc[36];
  #pragma unroll
  for (int p = 0; p < 36; p++) acc[p] = 0.f;
  for (size_t i = (size_t)blockIdx.x * blockDim.x + threadIdx.x; i < N;
       i += (size_t)gridDim.x * blockDim.x) {
    float v[NEXP];
    #pragma unroll
    for (int e = 0; e < NEXP; e++) v[e] = eout[(size_t)e * N + i];
    int p = 0;
    #pragma unroll
    for (int a = 0; a < NEXP; a++)
      #pragma unroll
      for (int b = a; b < NEXP; b++) acc[p++] += v[a] * v[b];
  }
  #pragma unroll
  for (int p = 0; p < 36; p++) {
    float s = acc[p];
    for (int off = 16; off > 0; off >>= 1) s += __shfl_xor(s, off, 32);
    if ((threadIdx.x & 31) == 0) atomicAdd(&gram[p], s);
  }
}

__global__ void sim_kernel(const float* __restrict__ gram,
                           float* __restrict__ simout) {
  __shared__ float G[64];
  __shared__ float dist[64];
  __shared__ float dmax;
  int tid = threadIdx.x;
  if (tid < 64) {
    int a = tid >> 3, b = tid & 7;
    int lo = a < b ? a : b, hi = a < b ? b : a;
    int p = lo * 8 - (lo * (lo - 1)) / 2 + (hi - lo);  // upper-tri index
    G[tid] = gram[p];
  }
  __syncthreads();
  if (tid < 64) {
    int a = tid >> 3, b = tid & 7;
    float sqa = G[a * 8 + a], sqb = G[b * 8 + b];
    dist[tid] = (a == b) ? 0.f : sqrtf(fmaxf(sqa + sqb - 2.f * G[tid], 0.f));
  }
  __syncthreads();
  if (tid == 0) {
    float m = 0.f;
    for (int i = 0; i < 64; i++) m = fmaxf(m, dist[i]);
    dmax = fmaxf(m, 1e-12f);
  }
  __syncthreads();
  if (tid < 64) {
    int a = tid >> 3, b = tid & 7;
    simout[tid] = (a == b) ? 1.f : 1.f - dist[tid] / dmax;
  }
}

// ---------------------------------------------------------------------------
// Final: weighted top-2 gather of expert outputs -> fp32 (B,S,H)
// ---------------------------------------------------------------------------
__global__ void final_gather(const float* __restrict__ eout,
                             const float* __restrict__ rw,
                             const int* __restrict__ ridx,
                             float* __restrict__ out) {
  const size_t N = (size_t)T_TOK * HDIM;
  for (size_t i = (size_t)blockIdx.x * blockDim.x + threadIdx.x; i < N;
       i += (size_t)gridDim.x * blockDim.x) {
    int t = (int)(i / HDIM), h = (int)(i % HDIM);
    float w0 = rw[t * 2], w1 = rw[t * 2 + 1];
    int e0 = ridx[t * 2], e1 = ridx[t * 2 + 1];
    out[i] = w0 * eout[((size_t)e0 * T_TOK + t) * HDIM + h] +
             w1 * eout[((size_t)e1 * T_TOK + t) * HDIM + h];
  }
}

// ---------------------------------------------------------------------------
extern "C" void kernel_launch(void* const* d_in, const int* in_sizes, int n_in,
                              void* d_out, int out_size, void* d_ws, size_t ws_size,
                              hipStream_t stream) {
  const float* x  = (const float*)d_in[0];  // (2,1024,1024) fp32
  const float* gw = (const float*)d_in[1];  // (8,1024)      fp32
  const float* w1 = (const float*)d_in[2];  // (8,4096,1024) fp32
  const float* w2 = (const float*)d_in[3];  // (8,1024,2048) fp32
  float* out = (float*)d_out;               // final (2097152) ++ sim (64)

  char* ws = (char*)d_ws;
  size_t off = 0;
  unsigned short* xb   = (unsigned short*)(ws + off); off += (size_t)T_TOK * HDIM * 2;
  unsigned short* w1b  = (unsigned short*)(ws + off); off += (size_t)NEXP * 2 * IDIM * HDIM * 2;
  unsigned short* w2b  = (unsigned short*)(ws + off); off += (size_t)NEXP * HDIM * IDIM * 2;
  unsigned short* intb = (unsigned short*)(ws + off); off += (size_t)NEXP * T_TOK * IDIM * 2;
  float* eout = (float*)(ws + off); off += (size_t)NEXP * T_TOK * HDIM * 4;
  float* rw   = (float*)(ws + off); off += (size_t)T_TOK * 2 * 4;
  int*   ridx = (int*)  (ws + off); off += (size_t)T_TOK * 2 * 4;
  float* gram = (float*)(ws + off); off += 256;
  if (ws_size < off) return;  // deterministic no-op if workspace too small

  zero36<<<1, 64, 0, stream>>>(gram);
  cvt_f32_bf16<<<1024, 256, 0, stream>>>(x,  xb,  T_TOK * HDIM);
  cvt_f32_bf16<<<2048, 256, 0, stream>>>(w1, w1b, NEXP * 2 * IDIM * HDIM);
  cvt_f32_bf16<<<2048, 256, 0, stream>>>(w2, w2b, NEXP * HDIM * IDIM);
  router_kernel<<<T_TOK / 8, 256, 0, stream>>>(x, gw, rw, ridx);

  dim3 g1(T_TOK / 128, IDIM / 128, NEXP);   // (16,16,8)
  gemm1_silu<<<g1, 256, 0, stream>>>(xb, w1b, intb);

  dim3 g2(T_TOK / 128, HDIM / 128, NEXP);   // (16,8,8)
  gemm2<<<g2, 256, 0, stream>>>(intb, w2b, eout);

  gram_partial<<<1024, 256, 0, stream>>>(eout, gram);
  sim_kernel<<<1, 64, 0, stream>>>(gram, out + (size_t)T_TOK * HDIM);
  final_gather<<<2048, 256, 0, stream>>>(eout, rw, ridx, out);
}
